// BipartiteGNN_44057774522846
// MI455X (gfx1250) — matrix operand order repaired and checked by reading, not compiled, not run on hardware
//
#include <hip/hip_runtime.h>
#include <hip/hip_bf16.h>

typedef __attribute__((ext_vector_type(16))) _Float16 v16h;
typedef __attribute__((ext_vector_type(8)))  _Float16 v8h;
typedef __attribute__((ext_vector_type(4)))  _Float16 h4;
typedef __attribute__((ext_vector_type(2)))  _Float16 h2;
typedef __attribute__((ext_vector_type(8)))  float    v8f;

#define N_COL 100000
#define N_CON 20000
#define NEDGE 800000
#define D_COL 16
#define D_CON 8
#define HDIM  128
#define KPAD  32

// ---------------------------------------------------------------------------
// Fragment loaders. Layout per ISA 7.12.2:
// A 16x32 f16: lane L holds row M=L&15, half=L>>4;
//   elem e<8  -> K = kc + half*8 + e ; elem e>=8 -> K = kc + 16 + half*8 + (e-8)
// B 32x16 f16: lane L holds col N=L&15, half=L>>4; same K mapping as A
// All f16 loads are 16B-aligned v8h (-> global_load_b128), zero conversion VALU.
// ---------------------------------------------------------------------------
__device__ __forceinline__ v16h load_a_frag16(const _Float16* __restrict__ rowbase,
                                              int kc, int half) {
  v8h q0 = ((const v8h*)(rowbase + kc + half * 8))[0];
  v8h q1 = ((const v8h*)(rowbase + kc + 16 + half * 8))[0];
  v16h a;
#pragma unroll
  for (int e = 0; e < 8; ++e) { a[e] = q0[e]; a[e + 8] = q1[e]; }
  return a;
}

__device__ __forceinline__ v16h load_b_frag16(const _Float16* __restrict__ colbase,
                                              int kc, int half) {
  v8h lo = ((const v8h*)(colbase + kc + half * 8))[0];
  v8h hi = ((const v8h*)(colbase + kc + 16 + half * 8))[0];
  v16h b;
#pragma unroll
  for (int e = 0; e < 8; ++e) { b[e] = lo[e]; b[e + 8] = hi[e]; }
  return b;
}

// f32 A fragment (agg buffer) with fused 1/deg row scale
__device__ __forceinline__ v16h load_a_frag32(const float* __restrict__ rowbase,
                                              int kc, int half, float scale) {
  const float4* p0 = (const float4*)(rowbase + kc + half * 8);
  const float4* p1 = (const float4*)(rowbase + kc + 16 + half * 8);
  float4 q0 = p0[0], q1 = p0[1];
  float4 q2 = p1[0], q3 = p1[1];
  v16h a;
  a[0]  = (_Float16)(q0.x * scale); a[1]  = (_Float16)(q0.y * scale);
  a[2]  = (_Float16)(q0.z * scale); a[3]  = (_Float16)(q0.w * scale);
  a[4]  = (_Float16)(q1.x * scale); a[5]  = (_Float16)(q1.y * scale);
  a[6]  = (_Float16)(q1.z * scale); a[7]  = (_Float16)(q1.w * scale);
  a[8]  = (_Float16)(q2.x * scale); a[9]  = (_Float16)(q2.y * scale);
  a[10] = (_Float16)(q2.z * scale); a[11] = (_Float16)(q2.w * scale);
  a[12] = (_Float16)(q3.x * scale); a[13] = (_Float16)(q3.y * scale);
  a[14] = (_Float16)(q3.z * scale); a[15] = (_Float16)(q3.w * scale);
  return a;
}

// C/D layout: lane L, VGPR v -> row = half*8 + v, col = L&15
// Epilogue: bias add (canonicalizes) + single max + cvt + 2B store.
template <int NT>
__device__ __forceinline__ void store_c_relu(v8f* acc, const float* __restrict__ bias,
                                             _Float16* __restrict__ out,
                                             int row_base, int half, int lcol) {
  const int N = NT * 16;
#pragma unroll
  for (int nt = 0; nt < NT; ++nt) {
    float bv = bias[nt * 16 + lcol];
#pragma unroll
    for (int v = 0; v < 8; ++v) {
      int r = row_base + half * 8 + v;
      float val = fmaxf(acc[nt][v] + bv, 0.f);
      out[(size_t)r * N + nt * 16 + lcol] = (_Float16)val;
    }
  }
}

// ---------------------------------------------------------------------------
// Activation GEMM: out[M x NT*16] = relu( A @ W^T + bias ), all-f16 operands.
// Block = 256 = 8 waves; wave w owns rows [blk*128 + 16w, +16) x all N.
// Zero-init accumulators let the first WMMA per tile use inline C=0.
// ---------------------------------------------------------------------------
template <int NT, int K>
__global__ void gemm_act(const _Float16* __restrict__ A, const _Float16* __restrict__ W,
                         const float* __restrict__ bias, _Float16* __restrict__ out, int M) {
  const int lane = threadIdx.x & 31;
  const int wave = threadIdx.x >> 5;
  const int half = lane >> 4;
  const int lcol = lane & 15;
  const int row_base = blockIdx.x * 128 + wave * 16;
  if (row_base >= M) return;  // wave-uniform (M multiple of 16)
  const int arow = row_base + lcol;

  const v8f zero = {};
  v8f acc[NT];
#pragma unroll
  for (int nt = 0; nt < NT; ++nt) acc[nt] = zero;

  const _Float16* abase = A + (size_t)arow * K;
#pragma unroll
  for (int kc = 0; kc < K; kc += 32) {
    v16h a = load_a_frag16(abase, kc, half);
#pragma unroll
    for (int nt = 0; nt < NT; ++nt) {
      v16h b = load_b_frag16(W + (size_t)(nt * 16 + lcol) * K, kc, half);
      acc[nt] = __builtin_amdgcn_wmma_f32_16x16x32_f16(false, a, false, b,
                                                       (short)0, acc[nt], false, false);
    }
  }
  store_c_relu<NT>(acc, bias, out, row_base, half, lcol);
}

// ---------------------------------------------------------------------------
// SAGE GEMM: out = relu( (A1f32/deg) @ W1^T + A2f16 @ W2^T + bias ), out f16.
// ---------------------------------------------------------------------------
template <int NT, int K>
__global__ void gemm_sage(const float* __restrict__ A1, const _Float16* __restrict__ W1,
                          const _Float16* __restrict__ A2, const _Float16* __restrict__ W2,
                          const float* __restrict__ bias, const float* __restrict__ deg,
                          _Float16* __restrict__ out, int M) {
  const int lane = threadIdx.x & 31;
  const int wave = threadIdx.x >> 5;
  const int half = lane >> 4;
  const int lcol = lane & 15;
  const int row_base = blockIdx.x * 128 + wave * 16;
  if (row_base >= M) return;
  const int arow = row_base + lcol;

  const float scale1 = 1.f / fmaxf(deg[arow], 1.f);

  const v8f zero = {};
  v8f acc[NT];
#pragma unroll
  for (int nt = 0; nt < NT; ++nt) acc[nt] = zero;

  const float* a1base = A1 + (size_t)arow * K;
#pragma unroll
  for (int kc = 0; kc < K; kc += 32) {
    v16h a = load_a_frag32(a1base, kc, half, scale1);
#pragma unroll
    for (int nt = 0; nt < NT; ++nt) {
      v16h b = load_b_frag16(W1 + (size_t)(nt * 16 + lcol) * K, kc, half);
      acc[nt] = __builtin_amdgcn_wmma_f32_16x16x32_f16(false, a, false, b,
                                                       (short)0, acc[nt], false, false);
    }
  }
  const _Float16* a2base = A2 + (size_t)arow * K;
#pragma unroll
  for (int kc = 0; kc < K; kc += 32) {
    v16h a = load_a_frag16(a2base, kc, half);
#pragma unroll
    for (int nt = 0; nt < NT; ++nt) {
      v16h b = load_b_frag16(W2 + (size_t)(nt * 16 + lcol) * K, kc, half);
      acc[nt] = __builtin_amdgcn_wmma_f32_16x16x32_f16(false, a, false, b,
                                                       (short)0, acc[nt], false, false);
    }
  }
  store_c_relu<NT>(acc, bias, out, row_base, half, lcol);
}

// ---------------------------------------------------------------------------
// Helpers: pad/convert, scatter, degree, fill, final dot.
// ---------------------------------------------------------------------------
__global__ void pad_rows32_f16(const float* __restrict__ src, _Float16* __restrict__ dst,
                               int M, int Din) {
  int i = (int)(blockIdx.x * (size_t)blockDim.x + threadIdx.x);
  int stride = (int)(gridDim.x * (size_t)blockDim.x);
  int total = M * KPAD;
  for (; i < total; i += stride) {
    int r = i >> 5, c = i & 31;
    dst[i] = (_Float16)((c < Din) ? src[r * Din + c] : 0.f);
  }
}

__global__ void cvt_f32_to_f16(const float* __restrict__ src, _Float16* __restrict__ dst, int n) {
  int i = (int)(blockIdx.x * (size_t)blockDim.x + threadIdx.x);
  int stride = (int)(gridDim.x * (size_t)blockDim.x);
  for (; i < n; i += stride) dst[i] = (_Float16)src[i];
}

// One wave per edge: gather 256B f16 row, scatter-add into f32 agg row.
__global__ void scatter_add_h128(const _Float16* __restrict__ src_feat,
                                 const int* __restrict__ src_idx,
                                 const int* __restrict__ dst_idx,
                                 float* __restrict__ agg, int E) {
  const int lane = threadIdx.x & 31;
  int wave = (int)((blockIdx.x * (size_t)blockDim.x + threadIdx.x) >> 5);
  const int nwaves = (int)((gridDim.x * (size_t)blockDim.x) >> 5);
  for (int e = wave; e < E; e += nwaves) {
    int s = src_idx[e];
    int d = dst_idx[e];
    h4 v = ((const h4*)(src_feat + (size_t)s * HDIM))[lane];  // 8B per lane
    float* dp = agg + (size_t)d * HDIM + lane * 4;
    __hip_atomic_fetch_add(dp + 0, (float)v[0], __ATOMIC_RELAXED, __HIP_MEMORY_SCOPE_AGENT);
    __hip_atomic_fetch_add(dp + 1, (float)v[1], __ATOMIC_RELAXED, __HIP_MEMORY_SCOPE_AGENT);
    __hip_atomic_fetch_add(dp + 2, (float)v[2], __ATOMIC_RELAXED, __HIP_MEMORY_SCOPE_AGENT);
    __hip_atomic_fetch_add(dp + 3, (float)v[3], __ATOMIC_RELAXED, __HIP_MEMORY_SCOPE_AGENT);
  }
}

__global__ void degree_count(const int* __restrict__ dst_idx, float* __restrict__ deg, int E) {
  int i = (int)(blockIdx.x * (size_t)blockDim.x + threadIdx.x);
  int stride = (int)(gridDim.x * (size_t)blockDim.x);
  for (; i < E; i += stride)
    __hip_atomic_fetch_add(&deg[dst_idx[i]], 1.f, __ATOMIC_RELAXED, __HIP_MEMORY_SCOPE_AGENT);
}

__global__ void fill_zero4(float4* __restrict__ p, size_t n4) {
  size_t i = blockIdx.x * (size_t)blockDim.x + threadIdx.x;
  size_t stride = (size_t)gridDim.x * blockDim.x;
  float4 z = {0.f, 0.f, 0.f, 0.f};
  for (; i < n4; i += stride) p[i] = z;
}

// out[r] = dot(hq[r][0:64], w2) + b2 ; wave per row, shfl butterfly reduce
__global__ void q_head2(const _Float16* __restrict__ hq, const float* __restrict__ w2,
                        const float* __restrict__ b2, float* __restrict__ out, int M) {
  const int lane = threadIdx.x & 31;
  int wave = (int)((blockIdx.x * (size_t)blockDim.x + threadIdx.x) >> 5);
  const int nwaves = (int)((gridDim.x * (size_t)blockDim.x) >> 5);
  float2 w = ((const float2*)w2)[lane];
  float bb = b2[0];
  for (int r = wave; r < M; r += nwaves) {
    h2 v = ((const h2*)(hq + (size_t)r * 64))[lane];
    float s = (float)v[0] * w.x + (float)v[1] * w.y;
#pragma unroll
    for (int off = 16; off > 0; off >>= 1) s += __shfl_xor(s, off, 32);
    if (lane == 0) out[r] = s + bb;
  }
}

// ---------------------------------------------------------------------------
extern "C" void kernel_launch(void* const* d_in, const int* in_sizes, int n_in,
                              void* d_out, int out_size, void* d_ws, size_t ws_size,
                              hipStream_t stream) {
  const float* x_col    = (const float*)d_in[0];
  const float* x_con    = (const float*)d_in[1];
  const int*   edge_col = (const int*)d_in[2];
  const int*   edge_con = (const int*)d_in[3];
  const float* W_col = (const float*)d_in[4];  const float* b_col = (const float*)d_in[5];
  const float* W_con = (const float*)d_in[6];  const float* b_con = (const float*)d_in[7];
  const float* c1cn_Wl = (const float*)d_in[8];  const float* c1cn_bl = (const float*)d_in[9];
  const float* c1cn_Wr = (const float*)d_in[10];
  const float* c1nc_Wl = (const float*)d_in[11]; const float* c1nc_bl = (const float*)d_in[12];
  const float* c1nc_Wr = (const float*)d_in[13];
  // d_in[14..16] = c2_cn_* : unused (h_con2 is dead in the reference)
  const float* c2nc_Wl = (const float*)d_in[17]; const float* c2nc_bl = (const float*)d_in[18];
  const float* c2nc_Wr = (const float*)d_in[19];
  const float* q_W1 = (const float*)d_in[20]; const float* q_b1 = (const float*)d_in[21];
  const float* q_W2 = (const float*)d_in[22]; const float* q_b2 = (const float*)d_in[23];
  float* out = (float*)d_out;

  // Workspace carve (256B aligned). Total ~121 MB.
  char* base = (char*)d_ws;
  size_t off = 0;
  auto carve = [&](size_t bytes) -> void* {
    off = (off + 255) & ~(size_t)255;
    void* p = base + off;
    off += bytes;
    return p;
  };
  _Float16* h_col = (_Float16*)carve((size_t)N_COL * HDIM * 2);  // later h_col2
  _Float16* h_con = (_Float16*)carve((size_t)N_CON * HDIM * 2);
  _Float16* n_col = (_Float16*)carve((size_t)N_COL * HDIM * 2);
  _Float16* n_con = (_Float16*)carve((size_t)N_CON * HDIM * 2);
  float*    agg   = (float*)carve((size_t)N_COL * HDIM * 4);     // f32 scatter target
  float*    deg_col = (float*)carve((size_t)N_COL * 4);
  float*    deg_con = (float*)carve((size_t)N_CON * 4);
  _Float16* xp_col = (_Float16*)carve((size_t)N_COL * KPAD * 2); // padded f16 inputs
  _Float16* xp_con = (_Float16*)carve((size_t)N_CON * KPAD * 2);
  _Float16* wp_col = (_Float16*)carve((size_t)HDIM * KPAD * 2);  // padded f16 weights
  _Float16* wp_con = (_Float16*)carve((size_t)HDIM * KPAD * 2);
  _Float16* w16 = (_Float16*)carve((size_t)7 * HDIM * HDIM * 2); // f16 weight pool
  _Float16* w_c1cn_Wl = w16 + 0 * HDIM * HDIM;
  _Float16* w_c1cn_Wr = w16 + 1 * HDIM * HDIM;
  _Float16* w_c1nc_Wl = w16 + 2 * HDIM * HDIM;
  _Float16* w_c1nc_Wr = w16 + 3 * HDIM * HDIM;
  _Float16* w_c2nc_Wl = w16 + 4 * HDIM * HDIM;
  _Float16* w_c2nc_Wr = w16 + 5 * HDIM * HDIM;
  _Float16* w_qW1     = w16 + 6 * HDIM * HDIM;   // 64x128 used
  _Float16* hq = (_Float16*)agg;  // [N_COL x 64] f16, reuses agg after conv2

  const dim3 blk(256);
  const int gemm_blocks_col = (N_COL + 127) / 128;  // 782
  const int gemm_blocks_con = (N_CON + 127) / 128;  // 157

  // Pad + convert embed operands; convert dense weights to f16
  pad_rows32_f16<<<2048, blk, 0, stream>>>(x_col, xp_col, N_COL, D_COL);
  pad_rows32_f16<<<512,  blk, 0, stream>>>(x_con, xp_con, N_CON, D_CON);
  pad_rows32_f16<<<8,    blk, 0, stream>>>(W_col, wp_col, HDIM, D_COL);
  pad_rows32_f16<<<8,    blk, 0, stream>>>(W_con, wp_con, HDIM, D_CON);
  cvt_f32_to_f16<<<16, blk, 0, stream>>>(c1cn_Wl, w_c1cn_Wl, HDIM * HDIM);
  cvt_f32_to_f16<<<16, blk, 0, stream>>>(c1cn_Wr, w_c1cn_Wr, HDIM * HDIM);
  cvt_f32_to_f16<<<16, blk, 0, stream>>>(c1nc_Wl, w_c1nc_Wl, HDIM * HDIM);
  cvt_f32_to_f16<<<16, blk, 0, stream>>>(c1nc_Wr, w_c1nc_Wr, HDIM * HDIM);
  cvt_f32_to_f16<<<16, blk, 0, stream>>>(c2nc_Wl, w_c2nc_Wl, HDIM * HDIM);
  cvt_f32_to_f16<<<16, blk, 0, stream>>>(c2nc_Wr, w_c2nc_Wr, HDIM * HDIM);
  cvt_f32_to_f16<<<16, blk, 0, stream>>>(q_W1, w_qW1, 64 * HDIM);

  // Degrees (same edge list for both convs)
  fill_zero4<<<256, blk, 0, stream>>>((float4*)deg_col, N_COL / 4);
  fill_zero4<<<64,  blk, 0, stream>>>((float4*)deg_con, N_CON / 4);
  degree_count<<<2048, blk, 0, stream>>>(edge_col, deg_col, NEDGE);
  degree_count<<<2048, blk, 0, stream>>>(edge_con, deg_con, NEDGE);

  // Input embeddings: h = relu(x @ W^T + b), K padded to 32
  gemm_act<8, KPAD><<<gemm_blocks_col, blk, 0, stream>>>(xp_col, wp_col, b_col, h_col, N_COL);
  gemm_act<8, KPAD><<<gemm_blocks_con, blk, 0, stream>>>(xp_con, wp_con, b_con, h_con, N_CON);

  // conv1 col->con: n_con = relu(mean_agg(h_col) @ Wl^T + bl + h_con @ Wr^T)
  fill_zero4<<<2048, blk, 0, stream>>>((float4*)agg, (size_t)N_CON * HDIM / 4);
  scatter_add_h128<<<8192, blk, 0, stream>>>(h_col, edge_col, edge_con, agg, NEDGE);
  gemm_sage<8, HDIM><<<gemm_blocks_con, blk, 0, stream>>>(
      agg, w_c1cn_Wl, h_con, w_c1cn_Wr, c1cn_bl, deg_con, n_con, N_CON);

  // conv1 con->col: n_col
  fill_zero4<<<2048, blk, 0, stream>>>((float4*)agg, (size_t)N_COL * HDIM / 4);
  scatter_add_h128<<<8192, blk, 0, stream>>>(h_con, edge_con, edge_col, agg, NEDGE);
  gemm_sage<8, HDIM><<<gemm_blocks_col, blk, 0, stream>>>(
      agg, w_c1nc_Wl, h_col, w_c1nc_Wr, c1nc_bl, deg_col, n_col, N_COL);

  // conv2 con->col only (h_con2 dead): h_col2 overwrites h_col
  fill_zero4<<<2048, blk, 0, stream>>>((float4*)agg, (size_t)N_COL * HDIM / 4);
  scatter_add_h128<<<8192, blk, 0, stream>>>(n_con, edge_con, edge_col, agg, NEDGE);
  gemm_sage<8, HDIM><<<gemm_blocks_col, blk, 0, stream>>>(
      agg, w_c2nc_Wl, n_col, w_c2nc_Wr, c2nc_bl, deg_col, h_col, N_COL);

  // Q head: hq = relu(h_col2 @ qW1^T + b1) [N_COL x 64]; out = hq @ qW2^T + b2
  gemm_act<4, HDIM><<<gemm_blocks_col, blk, 0, stream>>>(h_col, w_qW1, q_b1, hq, N_COL);
  q_head2<<<4096, blk, 0, stream>>>(hq, q_W2, q_b2, out, N_COL);
}
